// BaselineTransformerLayer_55155970015382
// MI455X (gfx1250) — matrix operand order repaired and checked by reading, not compiled, not run on hardware
//
#include <hip/hip_runtime.h>
#include <hip/hip_bf16.h>
#include <stdint.h>

#define SEQ   2048
#define BATCH 2
#define HIDN  2048
#define NH    16
#define NKV   4
#define HD    128
#define NEXP  8
#define FFN   4096
#define MROWS (SEQ*BATCH)          // 4096
#define QKVN  (NKV*((NH/NKV)+2)*HD) // 3072
#define GSZ   ((NH/NKV+2)*HD)       // 768

typedef __attribute__((ext_vector_type(16))) __bf16 v16bf;
typedef __attribute__((ext_vector_type(8)))  float  v8f;

union FragB { uint32_t u[8]; v16bf v; };

__device__ __forceinline__ uint16_t f2bf(float f) {
  union { __bf16 h; uint16_t u; } c; c.h = (__bf16)f; return c.u;
}
__device__ __forceinline__ float bf2f(uint16_t u16) {
  union { __bf16 h; uint16_t u; } c; c.u = u16; return (float)c.h;
}
// A-matrix (16x32 bf16) per-lane K offsets: half-lane 0 -> K {0..7,16..23}, half 1 -> {8..15,24..31}
__device__ __forceinline__ int frag_ka(int half, int v) {
  return (v < 4) ? (half*8 + 2*v) : (16 + half*8 + 2*(v-4));
}
// B-matrix (32x16 bf16) per-lane K offsets: half-lane 0 -> K {0..7,16..23}, half 1 -> {8..15,24..31}
__device__ __forceinline__ int frag_kb(int half, int v) {
  return (v < 4) ? (half*8 + 2*v) : (16 + half*8 + 2*(v-4));
}

// ---------------- LayerNorm -> bf16 (one wave per row) ----------------
__global__ void ln_bf16_kernel(const float* __restrict__ x, const float* __restrict__ w,
                               const float* __restrict__ b, uint16_t* __restrict__ out, int rows) {
  int wid = threadIdx.x >> 5, lane = threadIdx.x & 31;
  int row = blockIdx.x * 8 + wid;
  if (row >= rows) return;
  const float* xr = x + (size_t)row * HIDN;
  float s = 0.f, s2 = 0.f;
  for (int i = lane; i < HIDN; i += 32) { float v = xr[i]; s += v; s2 += v*v; }
  for (int d = 16; d; d >>= 1) { s += __shfl_xor(s, d, 32); s2 += __shfl_xor(s2, d, 32); }
  float mu  = s * (1.0f / HIDN);
  float var = s2 * (1.0f / HIDN) - mu * mu;
  float inv = rsqrtf(var + 1e-5f);
  uint16_t* orow = out + (size_t)row * HIDN;
  for (int i = lane; i < HIDN; i += 32)
    orow[i] = f2bf((xr[i] - mu) * inv * w[i] + b[i]);
}

// ---------------- f32 -> bf16 copy ----------------
__global__ void conv_bf16_kernel(const float* __restrict__ src, uint16_t* __restrict__ dst, size_t n) {
  size_t i = (size_t)blockIdx.x * blockDim.x + threadIdx.x;
  if (i < n) dst[i] = f2bf(src[i]);
}

// ---------------- (E,K,N) f32 -> (E,N,K) bf16 transpose ----------------
__global__ void convT_bf16_kernel(const float* __restrict__ src, uint16_t* __restrict__ dst, int K, int N) {
  size_t i = (size_t)blockIdx.x * blockDim.x + threadIdx.x;
  size_t per = (size_t)K * N;
  size_t total = (size_t)NEXP * per;
  if (i >= total) return;
  size_t e = i / per;
  size_t rem = i - e * per;
  int k = (int)(rem / N), n = (int)(rem % N);
  dst[e * per + (size_t)n * K + k] = f2bf(src[i]);
}

// ---------------- Tiled bf16 WMMA GEMM: C[M,N] = A[M,K] @ W[N,K]^T ----------------
// mode 0: Cf = v;  mode 1: Cf += v;  mode 3: Cb = bf16(gelu_exact(v));
// mode 4: Cf += rs[m*rs_stride] * v
#define BM 128
#define BN 128
#define BK 32
#define LDA 40   // padded LDS row stride (elements)

__global__ __launch_bounds__(256)
void gemm_bf16_kernel(const uint16_t* __restrict__ A, const uint16_t* __restrict__ W,
                      float* Cf, uint16_t* Cb, int M, int N, int K,
                      int mode, const float* __restrict__ rs, int rs_stride) {
  __shared__ uint16_t As[BM * LDA];
  __shared__ uint16_t Bs[BN * LDA];
  int tid = threadIdx.x, wid = tid >> 5, lane = tid & 31;
  int half = lane >> 4, l16 = lane & 15;
  int wm = (wid & 3) * 32;   // 4 waves along M
  int wn = (wid >> 2) * 64;  // 2 waves along N
  int m0 = blockIdx.y * BM, n0 = blockIdx.x * BN;

  v8f acc[2][4];
  for (int i = 0; i < 2; i++)
    for (int j = 0; j < 4; j++)
      for (int r = 0; r < 8; r++) acc[i][j][r] = 0.f;

  int ar = tid >> 1, ac = (tid & 1) * 16;  // each thread stages 16 bf16 (32B) per tile
  const uint16_t* gA = A + (size_t)(m0 + ar) * K + ac;
  const uint16_t* gB = W + (size_t)(n0 + ar) * K + ac;
  uint32_t ldsA = (uint32_t)(uintptr_t)(As + ar * LDA + ac); // wave-relative LDS offset
  uint32_t ldsB = (uint32_t)(uintptr_t)(Bs + ar * LDA + ac);

  for (int kb = 0; kb < K; kb += BK) {
    {
      // CDNA5 async copy: memory -> LDS without VGPR round trip (ASYNCcnt tracked).
      // INST_OFFSET applies to both global and LDS sides, so offset:16 covers beat 2.
      uint64_t ga = (uint64_t)(uintptr_t)(gA + kb);
      uint64_t gb = (uint64_t)(uintptr_t)(gB + kb);
      asm volatile(
        "global_load_async_to_lds_b128 %0, %2, off\n\t"
        "global_load_async_to_lds_b128 %0, %2, off offset:16\n\t"
        "global_load_async_to_lds_b128 %1, %3, off\n\t"
        "global_load_async_to_lds_b128 %1, %3, off offset:16"
        :
        : "v"(ldsA), "v"(ldsB), "v"(ga), "v"(gb)
        : "memory");
      if (kb + BK < K) { // prefetch next K tile of the streamed operands
        __builtin_prefetch(gA + kb + BK, 0, 1);
        __builtin_prefetch(gB + kb + BK, 0, 1);
      }
      asm volatile("s_wait_asynccnt 0x0" ::: "memory");
    }
    __syncthreads();
    FragB af[2], bfr[4];
    for (int mi = 0; mi < 2; mi++) {
      const uint16_t* p = As + (wm + mi*16 + l16) * LDA;
      for (int v = 0; v < 8; v++) af[mi].u[v] = *(const uint32_t*)(p + frag_ka(half, v));
    }
    for (int ni = 0; ni < 4; ni++) {
      const uint16_t* p = Bs + (wn + ni*16 + l16) * LDA;
      for (int v = 0; v < 8; v++) bfr[ni].u[v] = *(const uint32_t*)(p + frag_kb(half, v));
    }
    for (int mi = 0; mi < 2; mi++)
      for (int ni = 0; ni < 4; ni++)
        acc[mi][ni] = __builtin_amdgcn_wmma_f32_16x16x32_bf16(
            false, af[mi].v, false, bfr[ni].v, (short)0, acc[mi][ni], false, false);
    __syncthreads();
  }
  for (int mi = 0; mi < 2; mi++)
    for (int ni = 0; ni < 4; ni++)
      for (int r = 0; r < 8; r++) {
        int m = m0 + wm + mi*16 + r + half*8;
        int n = n0 + wn + ni*16 + l16;
        float v = acc[mi][ni][r];
        size_t idx = (size_t)m * N + n;
        if (mode == 0)      Cf[idx] = v;
        else if (mode == 1) Cf[idx] += v;
        else if (mode == 3) Cb[idx] = f2bf(0.5f * v * (1.0f + erff(v * 0.70710678118f)));
        else if (mode == 4) Cf[idx] += rs[(size_t)m * rs_stride] * v;
      }
}

// ---------------- qkv f32 -> bf16 Q (B,NH,S,HD), K (B,NKV,S,HD), V^T (B,NKV,HD,S) ----------------
__global__ void repack_qkv_kernel(const float* __restrict__ qkv, uint16_t* __restrict__ Qb,
                                  uint16_t* __restrict__ Kb, uint16_t* __restrict__ Vt) {
  size_t i = (size_t)blockIdx.x * blockDim.x + threadIdx.x;
  size_t total = (size_t)MROWS * QKVN;
  if (i >= total) return;
  int row = (int)(i / QKVN), col = (int)(i % QKVN);
  int s = row / BATCH, bb = row % BATCH;
  int g = col / GSZ, o = col % GSZ;
  uint16_t h16 = f2bf(qkv[i]);
  if (o < 4*HD) {
    int h = g*4 + o/HD, d = o % HD;
    Qb[(((size_t)bb*NH + h)*SEQ + s)*HD + d] = h16;
  } else if (o < 5*HD) {
    int d = o - 4*HD;
    Kb[(((size_t)bb*NKV + g)*SEQ + s)*HD + d] = h16;
  } else {
    int d = o - 5*HD;
    Vt[(((size_t)bb*NKV + g)*HD + d)*SEQ + s] = h16;
  }
}

// ---------------- Flash attention (causal, GQA) ----------------
#define KLD 136
#define VLD 40
#define PLD 40

__global__ __launch_bounds__(256)
void attn_kernel(const uint16_t* __restrict__ Qb, const uint16_t* __restrict__ Kb,
                 const uint16_t* __restrict__ Vt, uint16_t* __restrict__ out) {
  __shared__ uint16_t Ks[32 * KLD];     // keys x d
  __shared__ uint16_t Vs[HD * VLD];     // d x keys (pre-transposed V)
  __shared__ uint16_t Ps[8 * 16 * PLD]; // per-wave P staging
  int tid = threadIdx.x, wid = tid >> 5, lane = tid & 31;
  int half = lane >> 4, l16 = lane & 15;
  int bb = blockIdx.z, h = blockIdx.y, q0 = blockIdx.x * 128;
  int g = h / (NH / NKV);
  const uint16_t* Qh = Qb + ((size_t)(bb*NH + h)) * SEQ * HD;
  const uint16_t* Kh = Kb + ((size_t)(bb*NKV + g)) * SEQ * HD;
  const uint16_t* Vh = Vt + ((size_t)(bb*NKV + g)) * HD * SEQ;

  int qrow = q0 + wid*16 + l16;
  FragB qf[4];
  for (int ks = 0; ks < 4; ks++)
    for (int v = 0; v < 8; v++)
      qf[ks].u[v] = *(const uint32_t*)(Qh + (size_t)qrow * HD + ks*32 + frag_ka(half, v));

  v8f o[8];
  for (int i = 0; i < 8; i++) for (int r = 0; r < 8; r++) o[i][r] = 0.f;
  float mrun[8], lrun[8];
  for (int r = 0; r < 8; r++) { mrun[r] = -1e30f; lrun[r] = 0.f; }
  const float scale = 0.0883883476483f; // 1/sqrt(128)

  int jmax = (q0 + 127) / 32;
  for (int j = 0; j <= jmax; j++) {
    int j0 = j * 32;
    { // stage K (32x128) and V^T (128x32)
      int r = tid >> 3, seg = tid & 7;
      const float4* gk = (const float4*)(Kh + (size_t)(j0 + r) * HD + seg*16);
      float4* lk = (float4*)(Ks + r * KLD + seg*16);
      lk[0] = gk[0]; lk[1] = gk[1];
      int vr = tid >> 1, vh = tid & 1;
      const float4* gv = (const float4*)(Vh + (size_t)vr * SEQ + j0 + vh*16);
      float4* lv = (float4*)(Vs + vr * VLD + vh*16);
      lv[0] = gv[0]; lv[1] = gv[1];
    }
    __syncthreads();
    v8f sc[2];
    for (int t = 0; t < 2; t++) for (int r = 0; r < 8; r++) sc[t][r] = 0.f;
    for (int ks = 0; ks < 4; ks++)
      for (int t = 0; t < 2; t++) {
        FragB kf;
        const uint16_t* p = Ks + (t*16 + l16) * KLD + ks*32;
        for (int v = 0; v < 8; v++) kf.u[v] = *(const uint32_t*)(p + frag_kb(half, v));
        sc[t] = __builtin_amdgcn_wmma_f32_16x16x32_bf16(
            false, qf[ks].v, false, kf.v, (short)0, sc[t], false, false);
      }
    // online softmax: row stats live per (half-lane, r)
    int qi_base = q0 + wid*16 + half*8;
    for (int r = 0; r < 8; r++) {
      int qi = qi_base + r;
      for (int t = 0; t < 2; t++) {
        int col = j0 + t*16 + l16;
        float v = sc[t][r] * scale;
        if (col > qi) v = -1e9f;
        sc[t][r] = v;
      }
      float mx = fmaxf(sc[0][r], sc[1][r]);
      for (int d = 8; d; d >>= 1) mx = fmaxf(mx, __shfl_xor(mx, d, 32));
      float mnew = fmaxf(mrun[r], mx);
      float p0 = __expf(sc[0][r] - mnew), p1 = __expf(sc[1][r] - mnew);
      sc[0][r] = p0; sc[1][r] = p1;
      float rsum = p0 + p1;
      for (int d = 8; d; d >>= 1) rsum += __shfl_xor(rsum, d, 32);
      float alpha = __expf(mrun[r] - mnew);
      lrun[r] = lrun[r] * alpha + rsum;
      mrun[r] = mnew;
      for (int ni = 0; ni < 8; ni++) o[ni][r] *= alpha;
    }
    // P (D layout) -> LDS -> A fragment layout
    uint16_t* P = Ps + wid * 16 * PLD;
    for (int r = 0; r < 8; r++) {
      int row = r + half*8;
      P[row*PLD + l16]      = f2bf(sc[0][r]);
      P[row*PLD + 16 + l16] = f2bf(sc[1][r]);
    }
    __syncthreads();
    FragB pf;
    {
      const uint16_t* p = P + l16 * PLD;
      for (int v = 0; v < 8; v++) pf.u[v] = *(const uint32_t*)(p + frag_ka(half, v));
    }
    for (int ni = 0; ni < 8; ni++) {
      FragB vf;
      const uint16_t* p = Vs + (ni*16 + l16) * VLD;
      for (int v = 0; v < 8; v++) vf.u[v] = *(const uint32_t*)(p + frag_kb(half, v));
      o[ni] = __builtin_amdgcn_wmma_f32_16x16x32_bf16(
          false, pf.v, false, vf.v, (short)0, o[ni], false, false);
    }
    __syncthreads();
  }
  for (int r = 0; r < 8; r++) {
    int qi = q0 + wid*16 + half*8 + r;
    float inv = 1.0f / lrun[r];
    size_t rowbase = ((size_t)qi * BATCH + bb) * HIDN + (size_t)h * HD;
    for (int ni = 0; ni < 8; ni++)
      out[rowbase + ni*16 + l16] = f2bf(o[ni][r] * inv);
  }
}

// ---------------- router: logits -> softmax -> top2 -> dense gate[rows][E] ----------------
__global__ void router_kernel(const uint16_t* __restrict__ x, const float* __restrict__ Wr,
                              float* __restrict__ gate, int rows) {
  int wid = threadIdx.x >> 5, lane = threadIdx.x & 31;
  int row = blockIdx.x * 8 + wid;
  if (row >= rows) return;
  const uint16_t* xr = x + (size_t)row * HIDN;
  float acc[NEXP];
  for (int e = 0; e < NEXP; e++) acc[e] = 0.f;
  for (int i = lane; i < HIDN; i += 32) {
    float xv = bf2f(xr[i]);
    const float* wr = Wr + (size_t)i * NEXP;
    for (int e = 0; e < NEXP; e++) acc[e] += xv * wr[e];
  }
  for (int e = 0; e < NEXP; e++)
    for (int d = 16; d; d >>= 1) acc[e] += __shfl_xor(acc[e], d, 32);
  if (lane == 0) {
    float mx = acc[0];
    for (int e = 1; e < NEXP; e++) mx = fmaxf(mx, acc[e]);
    float p[NEXP]; float s = 0.f;
    for (int e = 0; e < NEXP; e++) { p[e] = __expf(acc[e] - mx); s += p[e]; }
    for (int e = 0; e < NEXP; e++) p[e] /= s;
    int i1 = 0;
    for (int e = 1; e < NEXP; e++) if (p[e] > p[i1]) i1 = e;
    int i2 = (i1 == 0) ? 1 : 0;
    for (int e = 0; e < NEXP; e++) if (e != i1 && p[e] > p[i2]) i2 = e;
    float denom = p[i1] + p[i2];
    for (int e = 0; e < NEXP; e++) gate[(size_t)row * NEXP + e] = 0.f;
    gate[(size_t)row * NEXP + i1] = p[i1] / denom;
    gate[(size_t)row * NEXP + i2] = p[i2] / denom;
  }
}

__global__ void add_kernel(const float* __restrict__ a, const float* __restrict__ b,
                           float* __restrict__ c, size_t n) {
  size_t i = (size_t)blockIdx.x * blockDim.x + threadIdx.x;
  if (i < n) c[i] = a[i] + b[i];
}

extern "C" void kernel_launch(void* const* d_in, const int* in_sizes, int n_in,
                              void* d_out, int out_size, void* d_ws, size_t ws_size,
                              hipStream_t stream) {
  const float* hidden  = (const float*)d_in[0];
  const float* ln1w    = (const float*)d_in[1];
  const float* ln1b    = (const float*)d_in[2];
  const float* ln2w    = (const float*)d_in[3];
  const float* ln2b    = (const float*)d_in[4];
  const float* qkvw    = (const float*)d_in[5];
  const float* projw   = (const float*)d_in[6];
  const float* routerw = (const float*)d_in[7];
  const float* w1      = (const float*)d_in[8];
  const float* w2      = (const float*)d_in[9];
  float* out = (float*)d_out;

  char* ws = (char*)d_ws;
  size_t off = 0;
  auto alloc = [&](size_t bytes) -> char* {
    char* p = ws + off;
    off = (off + bytes + 255) & ~(size_t)255;
    return p;
  };
  uint16_t* xln1   = (uint16_t*)alloc((size_t)MROWS * HIDN * 2);
  uint16_t* qkvwb  = (uint16_t*)alloc((size_t)QKVN * HIDN * 2);
  uint16_t* projwb = (uint16_t*)alloc((size_t)HIDN * HIDN * 2);
  uint16_t* w1T    = (uint16_t*)alloc((size_t)NEXP * FFN * HIDN * 2);
  uint16_t* w2T    = (uint16_t*)alloc((size_t)NEXP * HIDN * FFN * 2);
  float*    qkvo   = (float*)   alloc((size_t)MROWS * QKVN * 4);
  uint16_t* Qb     = (uint16_t*)alloc((size_t)BATCH * NH * SEQ * HD * 2);
  uint16_t* Kb     = (uint16_t*)alloc((size_t)BATCH * NKV * SEQ * HD * 2);
  uint16_t* Vt     = (uint16_t*)alloc((size_t)BATCH * NKV * HD * SEQ * 2);
  uint16_t* attnb  = (uint16_t*)alloc((size_t)MROWS * HIDN * 2);
  float*    projo  = (float*)   alloc((size_t)MROWS * HIDN * 4);
  float*    hafter = (float*)   alloc((size_t)MROWS * HIDN * 4);
  uint16_t* xln2   = (uint16_t*)alloc((size_t)MROWS * HIDN * 2);
  float*    gate   = (float*)   alloc((size_t)MROWS * NEXP * 4);
  uint16_t* hbuf   = (uint16_t*)alloc((size_t)MROWS * FFN * 2);
  float*    moeacc = (float*)   alloc((size_t)MROWS * HIDN * 4);
  (void)in_sizes; (void)n_in; (void)out_size; (void)ws_size;

  // LN1
  ln_bf16_kernel<<<MROWS/8, 256, 0, stream>>>(hidden, ln1w, ln1b, xln1, MROWS);
  // weight conversions
  {
    size_t n = (size_t)QKVN * HIDN;
    conv_bf16_kernel<<<(unsigned)((n + 255) / 256), 256, 0, stream>>>(qkvw, qkvwb, n);
    n = (size_t)HIDN * HIDN;
    conv_bf16_kernel<<<(unsigned)((n + 255) / 256), 256, 0, stream>>>(projw, projwb, n);
    n = (size_t)NEXP * HIDN * FFN;
    convT_bf16_kernel<<<(unsigned)((n + 255) / 256), 256, 0, stream>>>(w1, w1T, HIDN, FFN);
    convT_bf16_kernel<<<(unsigned)((n + 255) / 256), 256, 0, stream>>>(w2, w2T, FFN, HIDN);
  }
  // QKV GEMM: (4096,2048) @ (3072,2048)^T
  gemm_bf16_kernel<<<dim3(QKVN/128, MROWS/128), 256, 0, stream>>>(
      xln1, qkvwb, qkvo, nullptr, MROWS, QKVN, HIDN, 0, nullptr, 0);
  // repack Q/K/V (GQA handled by indexing; V pre-transposed d-major)
  {
    size_t n = (size_t)MROWS * QKVN;
    repack_qkv_kernel<<<(unsigned)((n + 255) / 256), 256, 0, stream>>>(qkvo, Qb, Kb, Vt);
  }
  // flash attention
  attn_kernel<<<dim3(SEQ/128, NH, BATCH), 256, 0, stream>>>(Qb, Kb, Vt, attnb);
  // output projection
  gemm_bf16_kernel<<<dim3(HIDN/128, MROWS/128), 256, 0, stream>>>(
      attnb, projwb, projo, nullptr, MROWS, HIDN, HIDN, 0, nullptr, 0);
  // residual 1
  {
    size_t n = (size_t)MROWS * HIDN;
    add_kernel<<<(unsigned)((n + 255) / 256), 256, 0, stream>>>(hidden, projo, hafter, n);
  }
  // LN2
  ln_bf16_kernel<<<MROWS/8, 256, 0, stream>>>(hafter, ln2w, ln2b, xln2, MROWS);
  // router top-2 gating
  router_kernel<<<MROWS/8, 256, 0, stream>>>(xln2, routerw, gate, MROWS);
  // MoE (dense per expert; gate folded into GEMM2 epilogue)
  hipMemsetAsync(moeacc, 0, (size_t)MROWS * HIDN * 4, stream);
  for (int e = 0; e < NEXP; e++) {
    gemm_bf16_kernel<<<dim3(FFN/128, MROWS/128), 256, 0, stream>>>(
        xln2, w1T + (size_t)e * FFN * HIDN, nullptr, hbuf,
        MROWS, FFN, HIDN, 3, nullptr, 0);                 // h = gelu(x @ w1[e]) in bf16
    gemm_bf16_kernel<<<dim3(HIDN/128, MROWS/128), 256, 0, stream>>>(
        hbuf, w2T + (size_t)e * HIDN * FFN, moeacc, nullptr,
        MROWS, HIDN, FFN, 4, gate + e, NEXP);             // moe += gate[:,e] * (h @ w2[e])
  }
  // final residual
  {
    size_t n = (size_t)MROWS * HIDN;
    add_kernel<<<(unsigned)((n + 255) / 256), 256, 0, stream>>>(hafter, moeacc, out, n);
  }
}
